// EGNNDynamics_53644141527276
// MI455X (gfx1250) — matrix operand order repaired and checked by reading, not compiled, not run on hardware
//
#include <hip/hip_runtime.h>
#include <hip/hip_bf16.h>

// ---------------- problem constants ----------------
#define BSZ   16
#define NPT   96
#define HDIM  128
#define NLAY  4
#define NSUB  2
#define NDIM  3
#define HFEAT 6
#define FINC  7      // HFEAT + 1
#define NFDIV 100.0f
#define JSPLIT 3     // j-range split for occupancy (6 j-tiles / 2 per block)

typedef __attribute__((ext_vector_type(16))) __bf16 v16bf;
typedef __attribute__((ext_vector_type(8)))  __bf16 v8bf;
typedef __attribute__((ext_vector_type(8)))  float  v8f;

static __device__ __forceinline__ float siluf(float x) {
    return x / (1.0f + __expf(-x));
}
static __device__ __forceinline__ __bf16 tobf(float x) { return (__bf16)x; }

// Build a v16bf WMMA fragment from two contiguous 8-element LDS runs.
static __device__ __forceinline__ v16bf frag16(const __bf16* p0, const __bf16* p1) {
    v8bf lo = *(const v8bf*)p0;
    v8bf hi = *(const v8bf*)p1;
    v16bf f;
#pragma unroll
    for (int q = 0; q < 8; ++q) { f[q] = lo[q]; f[8 + q] = hi[q]; }
    return f;
}

// Vectorized build of one 128-wide silu row into bf16 LDS (pair-MLP hidden act)
static __device__ __forceinline__ void build_mrow(__bf16* mrow, const float* hir,
                                                  const float* hjr, const float* wrb,
                                                  const float* wdb, const float* b1b,
                                                  float rr_, float dd) {
    const float4* h4 = (const float4*)hir;
    const float4* j4 = (const float4*)hjr;
    const float4* wr4 = (const float4*)wrb;
    const float4* wd4 = (const float4*)wdb;
    const float4* b14 = (const float4*)b1b;
#pragma unroll
    for (int c8 = 0; c8 < 16; ++c8) {
        v8bf pk;
#pragma unroll
        for (int q2 = 0; q2 < 2; ++q2) {
            int c4 = c8 * 2 + q2;
            float4 a = h4[c4], b = j4[c4], wr = wr4[c4], wd = wd4[c4], bb = b14[c4];
            pk[q2 * 4 + 0] = tobf(siluf(a.x + b.x + rr_ * wr.x + dd * wd.x + bb.x));
            pk[q2 * 4 + 1] = tobf(siluf(a.y + b.y + rr_ * wr.y + dd * wd.y + bb.y));
            pk[q2 * 4 + 2] = tobf(siluf(a.z + b.z + rr_ * wr.z + dd * wd.z + bb.z));
            pk[q2 * 4 + 3] = tobf(siluf(a.w + b.w + rr_ * wr.w + dd * wd.w + bb.w));
        }
        *(v8bf*)(mrow + c8 * 8) = pk;
    }
}

// =====================================================================
// Small elementwise kernels
// =====================================================================
__global__ void prep_h_kernel(const float* __restrict__ xh, const float* __restrict__ t,
                              const float* __restrict__ nm, const float* __restrict__ W_emb,
                              const float* __restrict__ b_emb, float* __restrict__ h) {
    int tid = blockIdx.x * 256 + threadIdx.x;          // BSZ*NPT*HDIM
    if (tid >= BSZ * NPT * HDIM) return;
    int r = tid >> 7, c = tid & 127;
    float m = nm[r];
    const float* row = xh + r * (NDIM + HFEAT);
    float acc = b_emb[c];
#pragma unroll
    for (int k = 0; k < HFEAT; ++k) acc += row[NDIM + k] * m * W_emb[k * HDIM + c];
    acc += t[0] * W_emb[HFEAT * HDIM + c];
    h[r * HDIM + c] = acc;
}

__global__ void prep_x_kernel(const float* __restrict__ xh, const float* __restrict__ nm,
                              float* __restrict__ x0, float* __restrict__ x) {
    int tid = blockIdx.x * 256 + threadIdx.x;          // BSZ*NPT*3
    if (tid >= BSZ * NPT * 3) return;
    int r = tid / 3, d = tid - r * 3;
    float v = xh[r * (NDIM + HFEAT) + d] * nm[r];
    x0[tid] = v; x[tid] = v;
}

__global__ void d0_kernel(const float* __restrict__ x0, float* __restrict__ d0) {
    int tid = blockIdx.x * 256 + threadIdx.x;          // BSZ*NPT*NPT
    if (tid >= BSZ * NPT * NPT) return;
    int j = tid % NPT; int bi = tid / NPT; int b = bi / NPT; int i = bi % NPT;
    const float* xi = x0 + (b * NPT + i) * 3;
    const float* xj = x0 + (b * NPT + j) * 3;
    float s = 0.f;
#pragma unroll
    for (int d = 0; d < 3; ++d) { float df = xi[d] - xj[d]; s += df * df; }
    d0[tid] = s;
}

__global__ void radial_kernel(const float* __restrict__ x, float* __restrict__ radial,
                              float* __restrict__ cdiff) {
    int tid = blockIdx.x * 256 + threadIdx.x;          // BSZ*NPT*NPT
    if (tid >= BSZ * NPT * NPT) return;
    int j = tid % NPT; int bi = tid / NPT; int b = bi / NPT; int i = bi % NPT;
    const float* xi = x + (b * NPT + i) * 3;
    const float* xj = x + (b * NPT + j) * 3;
    float df[3]; float s = 0.f;
#pragma unroll
    for (int d = 0; d < 3; ++d) { df[d] = xi[d] - xj[d]; s += df[d] * df[d]; }
    float inv = rsqrtf(s + 1e-8f);
    radial[tid] = s;
#pragma unroll
    for (int d = 0; d < 3; ++d) cdiff[tid * 3 + d] = df[d] * inv;
}

__global__ void xupdate_kernel(const float* __restrict__ xdelta, const float* __restrict__ nm,
                               float* __restrict__ x) {
    int tid = blockIdx.x * 256 + threadIdx.x;          // BSZ*NPT*3
    if (tid >= BSZ * NPT * 3) return;
    int r = tid / 3;
    x[tid] = (x[tid] + xdelta[tid] * (1.0f / NFDIV)) * nm[r];
}

__global__ void zero_kernel(float* __restrict__ p, int n) {
    int tid = blockIdx.x * 256 + threadIdx.x;
    if (tid < n) p[tid] = 0.0f;
}

__global__ void vel_kernel(const float* __restrict__ x, const float* __restrict__ x0,
                           const float* __restrict__ nm, float* __restrict__ vel,
                           float* __restrict__ velsum, float* __restrict__ nsum) {
    int r = blockIdx.x * 256 + threadIdx.x;            // BSZ*NPT
    if (r >= BSZ * NPT) return;
    int b = r / NPT;
    float m = nm[r];
#pragma unroll
    for (int d = 0; d < 3; ++d) {
        float v = (x[r * 3 + d] - x0[r * 3 + d]) * m;
        vel[r * 3 + d] = v;
        atomicAdd(&velsum[b * 3 + d], v);
    }
    atomicAdd(&nsum[b], m);
}

__global__ void final_kernel(const float* __restrict__ vel, const float* __restrict__ velsum,
                             const float* __restrict__ nsum, const float* __restrict__ h,
                             const float* __restrict__ W_out, const float* __restrict__ b_out,
                             const float* __restrict__ nm, float* __restrict__ out) {
    int r = blockIdx.x * 256 + threadIdx.x;            // BSZ*NPT
    if (r >= BSZ * NPT) return;
    int b = r / NPT;
    float m = nm[r];
    float inv = 1.0f / nsum[b];
#pragma unroll
    for (int d = 0; d < 3; ++d)
        out[r * (NDIM + HFEAT) + d] = vel[r * 3 + d] - velsum[b * 3 + d] * inv * m;
    for (int c = 0; c < HFEAT; ++c) {
        float acc = b_out[c];
#pragma unroll 8
        for (int k = 0; k < HDIM; ++k) acc += h[r * HDIM + k] * W_out[k * FINC + c];
        out[r * (NDIM + HFEAT) + NDIM + c] = acc * m;
    }
}

// =====================================================================
// Generic 1536-row GEMM (cols=128, K = K1[+K2], bf16 WMMA, f32 acc)
// grid = 96 blocks of 256 threads; wave w handles column tile w.
// =====================================================================
__global__ __launch_bounds__(256)
void gemm_rows_kernel(const void* __restrict__ A1, int K1, int a1bf,
                      const float* __restrict__ A2, int K2,
                      const float* __restrict__ B1, const float* __restrict__ B2,
                      const float* __restrict__ bias, const float* __restrict__ resid,
                      const float* __restrict__ rowmask, int do_silu,
                      float* __restrict__ outF, __bf16* __restrict__ outB) {
    const int tid = threadIdx.x;
    const int lane = tid & 31;
    const int w = tid >> 5;
    const int r0 = blockIdx.x * 16;
    const int K = K1 + K2;
    const int ldk = K + 8;                 // keeps 16B alignment (K multiple of 32)

    extern __shared__ char smem[];
    __bf16* abuf = (__bf16*)smem;          // 16 x ldk
    __bf16* wbuf = abuf + 16 * ldk;        // 128 x ldk (column-major weights)

    for (int e = tid; e < K * 128; e += 256) {
        int k = e >> 7, c = e & 127;
        float v = (k < K1) ? B1[k * 128 + c] : B2[(k - K1) * 128 + c];
        wbuf[c * ldk + k] = tobf(v);
    }
    for (int e = tid; e < 16 * K; e += 256) {
        int rr = e / K, k = e - rr * K;
        float v;
        if (k < K1) v = a1bf ? (float)((const __bf16*)A1)[(r0 + rr) * K1 + k]
                             : ((const float*)A1)[(r0 + rr) * K1 + k];
        else        v = A2[(r0 + rr) * K2 + (k - K1)];
        abuf[rr * ldk + k] = tobf(v);
    }
    __syncthreads();

    const int nkt = K >> 5;
    v8f acc = {};
    const __bf16* arow = abuf + (lane & 15) * ldk + ((lane < 16) ? 0 : 8);
    const __bf16* bcol = wbuf + (w * 16 + (lane & 15)) * ldk + ((lane < 16) ? 0 : 8);
    for (int kt = 0; kt < nkt; ++kt) {
        v16bf af = frag16(arow + kt * 32, arow + kt * 32 + 16);
        v16bf bf = frag16(bcol + kt * 32, bcol + kt * 32 + 16);
        acc = __builtin_amdgcn_wmma_f32_16x16x32_bf16(false, af, false, bf,
                                                      (short)0, acc, false, false);
    }

    const int col = w * 16 + (lane & 15);
    const int rowoff = 8 * (lane >> 4);
#pragma unroll
    for (int v = 0; v < 8; ++v) {
        int r = r0 + v + rowoff;
        float val = acc[v];
        if (bias)    val += bias[col];
        if (do_silu) val = siluf(val);
        if (resid)   val += resid[r * HDIM + col];
        if (rowmask) val *= rowmask[r];
        if (outF) outF[r * HDIM + col] = val;
        if (outB) outB[r * HDIM + col] = tobf(val);
    }
}

// =====================================================================
// Fused edge-message kernel: for one (b, i-tile, j-chunk) accumulate
// agg[i,:] += sum_j silu( silu(hi+hj+ea.W1e+b1) @ W2 + b2 ) * emask / NF
// grid = BSZ * (NPT/16) * JSPLIT = 288 blocks, 256 threads (8 waves).
// =====================================================================
__global__ __launch_bounds__(256)
void edge_fused_kernel(const float* __restrict__ hi, const float* __restrict__ hj,
                       const float* __restrict__ radial, const float* __restrict__ d0,
                       const float* __restrict__ emask, const float* __restrict__ W1ea,
                       const float* __restrict__ b1, const float* __restrict__ W2,
                       const float* __restrict__ b2, float* __restrict__ agg) {
    const int tid = threadIdx.x;
    const int lane = tid & 31;
    const int w = tid >> 5;
    const int b = blockIdx.x / (6 * JSPLIT);
    const int i0 = ((blockIdx.x / JSPLIT) % 6) * 16;
    const int js = blockIdx.x % JSPLIT;
    const int jt0 = js * (6 / JSPLIT), jt1 = jt0 + (6 / JSPLIT);

    extern __shared__ char smem[];
    __bf16* mbuf = (__bf16*)smem;              // 256 x 136 (pair rows x K)
    __bf16* wbuf = mbuf + 256 * 136;           // 128 x 136 (W2 col-major)
    float*  hib  = (float*)(wbuf + 128 * 136); // 16 x 128
    float*  hjb  = hib + 16 * 128;             // 16 x 128
    float*  wrb  = hjb + 16 * 128;             // 128
    float*  wdb  = wrb + 128;                  // 128
    float*  b1b  = wdb + 128;                  // 128
    float*  b2b  = b1b + 128;                  // 128

    for (int e = tid; e < 128 * 128; e += 256) {
        int k = e >> 7, c = e & 127;
        wbuf[c * 136 + k] = tobf(W2[e]);
    }
    for (int e = tid; e < 16 * 128; e += 256)
        hib[e] = hi[(b * NPT + i0 + (e >> 7)) * HDIM + (e & 127)];
    if (tid < 128) {
        wrb[tid] = W1ea[tid];
        wdb[tid] = W1ea[128 + tid];
        b1b[tid] = b1[tid];
        b2b[tid] = b2[tid];
    }
    // prefetch first hj chunk while staging weights
    __builtin_prefetch(hj + (b * NPT + jt0 * 16 + (tid >> 4)) * HDIM + (tid & 15) * 8, 0, 1);

    float accv[16];
#pragma unroll
    for (int q = 0; q < 16; ++q) accv[q] = 0.0f;

    for (int jt = jt0; jt < jt1; ++jt) {
        const int j0 = jt * 16;
        __syncthreads();
        for (int e = tid; e < 16 * 128; e += 256)
            hjb[e] = hj[(b * NPT + j0 + (e >> 7)) * HDIM + (e & 127)];
        if (jt + 1 < jt1)   // overlap next chunk fetch with this chunk's math
            __builtin_prefetch(hj + (b * NPT + (jt + 1) * 16 + (tid >> 4)) * HDIM +
                               (tid & 15) * 8, 0, 1);
        __syncthreads();
        {   // one pair row per thread
            int il = tid >> 4, jl = tid & 15;
            int i = i0 + il, j = j0 + jl;
            float rr_ = radial[(b * NPT + i) * NPT + j];
            float dd = d0[(b * NPT + i) * NPT + j];
            build_mrow(mbuf + tid * 136, hib + il * HDIM, hjb + jl * HDIM,
                       wrb, wdb, b1b, rr_, dd);
        }
        __syncthreads();
        // load A fragments for BOTH row tiles up front; reuse each B fragment twice
        v16bf afr[2][4];
        float em[2][8];
#pragma unroll
        for (int rr = 0; rr < 2; ++rr) {
            const int rt = w + 8 * rr;
            const int i = i0 + rt;
            const __bf16* arow = mbuf + (rt * 16 + (lane & 15)) * 136 + ((lane < 16) ? 0 : 8);
#pragma unroll
            for (int kt = 0; kt < 4; ++kt)
                afr[rr][kt] = frag16(arow + kt * 32, arow + kt * 32 + 16);
#pragma unroll
            for (int v = 0; v < 8; ++v)
                em[rr][v] = emask[(b * NPT + i) * NPT + j0 + v + 8 * (lane >> 4)];
        }
#pragma unroll
        for (int nt = 0; nt < 8; ++nt) {
            v8f acc0 = {}, acc1 = {};
            const __bf16* bcol = wbuf + (nt * 16 + (lane & 15)) * 136 + ((lane < 16) ? 0 : 8);
#pragma unroll
            for (int kt = 0; kt < 4; ++kt) {
                v16bf bfr = frag16(bcol + kt * 32, bcol + kt * 32 + 16);
                acc0 = __builtin_amdgcn_wmma_f32_16x16x32_bf16(false, afr[0][kt], false, bfr,
                                                               (short)0, acc0, false, false);
                acc1 = __builtin_amdgcn_wmma_f32_16x16x32_bf16(false, afr[1][kt], false, bfr,
                                                               (short)0, acc1, false, false);
            }
            const int col = nt * 16 + (lane & 15);
            float p0 = 0.0f, p1 = 0.0f;
#pragma unroll
            for (int v = 0; v < 8; ++v) {
                p0 += siluf(acc0[v] + b2b[col]) * em[0][v];
                p1 += siluf(acc1[v] + b2b[col]) * em[1][v];
            }
            accv[nt] += p0;
            accv[8 + nt] += p1;
        }
    }
#pragma unroll
    for (int rr = 0; rr < 2; ++rr) {
        const int rt = w + 8 * rr;
#pragma unroll
        for (int nt = 0; nt < 8; ++nt) {
            float s = accv[rr * 8 + nt];
            s += __shfl_xor(s, 16);
            if (lane < 16)
                atomicAdd(&agg[(b * NPT + i0 + rt) * HDIM + nt * 16 + lane],
                          s * (1.0f / NFDIV));
        }
    }
}

// =====================================================================
// Fused coordinate kernel: phi = silu(silu(hi+hj+ea.Wc1+b1) @ Wc2 + b2),
// s = phi @ w3;  xdelta[i,:] += sum_j cdiff[i,j,:] * s * emask
// grid = 288 blocks (same decode as edge kernel)
// =====================================================================
__global__ __launch_bounds__(256)
void coord_fused_kernel(const float* __restrict__ hi, const float* __restrict__ hj,
                        const float* __restrict__ radial, const float* __restrict__ d0,
                        const float* __restrict__ emask, const float* __restrict__ cdiff,
                        const float* __restrict__ W1ea, const float* __restrict__ b1,
                        const float* __restrict__ W2, const float* __restrict__ b2,
                        const float* __restrict__ w3, float* __restrict__ xdelta) {
    const int tid = threadIdx.x;
    const int lane = tid & 31;
    const int w = tid >> 5;
    const int b = blockIdx.x / (6 * JSPLIT);
    const int i0 = ((blockIdx.x / JSPLIT) % 6) * 16;
    const int js = blockIdx.x % JSPLIT;
    const int jt0 = js * (6 / JSPLIT), jt1 = jt0 + (6 / JSPLIT);

    extern __shared__ char smem[];
    __bf16* mbuf = (__bf16*)smem;
    __bf16* wbuf = mbuf + 256 * 136;
    float*  hib  = (float*)(wbuf + 128 * 136);
    float*  hjb  = hib + 16 * 128;
    float*  wrb  = hjb + 16 * 128;
    float*  wdb  = wrb + 128;
    float*  b1b  = wdb + 128;
    float*  b2b  = b1b + 128;
    float*  w3b  = b2b + 128;                  // 128
    float*  xaccL = w3b + 128;                 // 16 x 3

    for (int e = tid; e < 128 * 128; e += 256) {
        int k = e >> 7, c = e & 127;
        wbuf[c * 136 + k] = tobf(W2[e]);
    }
    for (int e = tid; e < 16 * 128; e += 256)
        hib[e] = hi[(b * NPT + i0 + (e >> 7)) * HDIM + (e & 127)];
    if (tid < 128) {
        wrb[tid] = W1ea[tid];
        wdb[tid] = W1ea[128 + tid];
        b1b[tid] = b1[tid];
        b2b[tid] = b2[tid];
        w3b[tid] = w3[tid];
    }
    if (tid < 48) xaccL[tid] = 0.0f;
    __builtin_prefetch(hj + (b * NPT + jt0 * 16 + (tid >> 4)) * HDIM + (tid & 15) * 8, 0, 1);

    for (int jt = jt0; jt < jt1; ++jt) {
        const int j0 = jt * 16;
        __syncthreads();
        for (int e = tid; e < 16 * 128; e += 256)
            hjb[e] = hj[(b * NPT + j0 + (e >> 7)) * HDIM + (e & 127)];
        if (jt + 1 < jt1)
            __builtin_prefetch(hj + (b * NPT + (jt + 1) * 16 + (tid >> 4)) * HDIM +
                               (tid & 15) * 8, 0, 1);
        __syncthreads();
        {
            int il = tid >> 4, jl = tid & 15;
            int i = i0 + il, j = j0 + jl;
            float rr_ = radial[(b * NPT + i) * NPT + j];
            float dd = d0[(b * NPT + i) * NPT + j];
            build_mrow(mbuf + tid * 136, hib + il * HDIM, hjb + jl * HDIM,
                       wrb, wdb, b1b, rr_, dd);
        }
        __syncthreads();
        v16bf afr[2][4];
#pragma unroll
        for (int rr = 0; rr < 2; ++rr) {
            const int rt = w + 8 * rr;
            const __bf16* arow = mbuf + (rt * 16 + (lane & 15)) * 136 + ((lane < 16) ? 0 : 8);
#pragma unroll
            for (int kt = 0; kt < 4; ++kt)
                afr[rr][kt] = frag16(arow + kt * 32, arow + kt * 32 + 16);
        }
        float pacc[2][8];
#pragma unroll
        for (int rr = 0; rr < 2; ++rr)
#pragma unroll
            for (int v = 0; v < 8; ++v) pacc[rr][v] = 0.0f;
#pragma unroll
        for (int nt = 0; nt < 8; ++nt) {
            v8f acc0 = {}, acc1 = {};
            const __bf16* bcol = wbuf + (nt * 16 + (lane & 15)) * 136 + ((lane < 16) ? 0 : 8);
#pragma unroll
            for (int kt = 0; kt < 4; ++kt) {
                v16bf bfr = frag16(bcol + kt * 32, bcol + kt * 32 + 16);
                acc0 = __builtin_amdgcn_wmma_f32_16x16x32_bf16(false, afr[0][kt], false, bfr,
                                                               (short)0, acc0, false, false);
                acc1 = __builtin_amdgcn_wmma_f32_16x16x32_bf16(false, afr[1][kt], false, bfr,
                                                               (short)0, acc1, false, false);
            }
            const int col = nt * 16 + (lane & 15);
#pragma unroll
            for (int v = 0; v < 8; ++v) {
                pacc[0][v] += siluf(acc0[v] + b2b[col]) * w3b[col];
                pacc[1][v] += siluf(acc1[v] + b2b[col]) * w3b[col];
            }
        }
        // reduce phi.w3 across the 16 lanes that share each row
#pragma unroll
        for (int off = 1; off < 16; off <<= 1)
#pragma unroll
            for (int rr = 0; rr < 2; ++rr)
#pragma unroll
                for (int v = 0; v < 8; ++v) pacc[rr][v] += __shfl_xor(pacc[rr][v], off);
        if ((lane & 15) == 0) {
            const int half = lane >> 4;
#pragma unroll
            for (int rr = 0; rr < 2; ++rr) {
                const int rt = w + 8 * rr;
                const int i = i0 + rt;
#pragma unroll
                for (int v = 0; v < 8; ++v) {
                    int j = j0 + v + 8 * half;
                    float s = pacc[rr][v] * emask[(b * NPT + i) * NPT + j];
                    const float* cd = cdiff + ((size_t)(b * NPT + i) * NPT + j) * 3;
                    atomicAdd(&xaccL[rt * 3 + 0], s * cd[0]);
                    atomicAdd(&xaccL[rt * 3 + 1], s * cd[1]);
                    atomicAdd(&xaccL[rt * 3 + 2], s * cd[2]);
                }
            }
        }
    }
    __syncthreads();
    if (tid < 48)
        atomicAdd(&xdelta[(b * NPT + i0 + tid / 3) * 3 + (tid % 3)], xaccL[tid]);
}

// =====================================================================
// Host orchestration
// =====================================================================
extern "C" void kernel_launch(void* const* d_in, const int* in_sizes, int n_in,
                              void* d_out, int out_size, void* d_ws, size_t ws_size,
                              hipStream_t stream) {
    const float* xh    = (const float*)d_in[0];
    const float* t     = (const float*)d_in[1];
    const float* nm    = (const float*)d_in[2];
    const float* em    = (const float*)d_in[3];
    const float* W_emb = (const float*)d_in[4];
    const float* b_emb = (const float*)d_in[5];
    const float* gWe1  = (const float*)d_in[6];
    const float* gbe1  = (const float*)d_in[7];
    const float* gWe2  = (const float*)d_in[8];
    const float* gbe2  = (const float*)d_in[9];
    const float* gWn1  = (const float*)d_in[10];
    const float* gbn1  = (const float*)d_in[11];
    const float* gWn2  = (const float*)d_in[12];
    const float* gbn2  = (const float*)d_in[13];
    const float* eWc1  = (const float*)d_in[14];
    const float* ebc1  = (const float*)d_in[15];
    const float* eWc2  = (const float*)d_in[16];
    const float* ebc2  = (const float*)d_in[17];
    const float* eWc3  = (const float*)d_in[18];
    const float* W_out = (const float*)d_in[19];
    const float* b_out = (const float*)d_in[20];

    char* wp = (char*)d_ws;
    auto alloc = [&](size_t bytes) { char* p = wp; wp += (bytes + 255) & ~size_t(255); return p; };
    const size_t NH = (size_t)BSZ * NPT * HDIM;
    float*  h       = (float*)alloc(NH * 4);
    float*  hi      = (float*)alloc(NH * 4);
    float*  hj      = (float*)alloc(NH * 4);
    float*  agg     = (float*)alloc(NH * 4);
    __bf16* ubf     = (__bf16*)alloc(NH * 2);
    float*  x0      = (float*)alloc((size_t)BSZ * NPT * 3 * 4);
    float*  x       = (float*)alloc((size_t)BSZ * NPT * 3 * 4);
    float*  xdelta  = (float*)alloc((size_t)BSZ * NPT * 3 * 4);
    float*  d0b     = (float*)alloc((size_t)BSZ * NPT * NPT * 4);
    float*  radb    = (float*)alloc((size_t)BSZ * NPT * NPT * 4);
    float*  cdiffb  = (float*)alloc((size_t)BSZ * NPT * NPT * 3 * 4);
    float*  vel     = (float*)alloc((size_t)BSZ * NPT * 3 * 4);
    float*  velsum  = (float*)alloc(64 * 4);   // [48] velsum + [16] nsum
    float*  nsum    = velsum + 48;

    const size_t shGemm128 = (size_t)(16 + 128) * (128 + 8) * 2;
    const size_t shGemm256 = (size_t)(16 + 128) * (256 + 8) * 2;
    const size_t shEdge    = (size_t)(256 + 128) * 136 * 2 + (16 * 128 * 2 + 4 * 128) * 4;
    const size_t shCoord   = shEdge + (128 + 48) * 4;
    const int    edgeGrid  = BSZ * 6 * JSPLIT;   // 288

    prep_h_kernel<<<768, 256, 0, stream>>>(xh, t, nm, W_emb, b_emb, h);
    prep_x_kernel<<<18, 256, 0, stream>>>(xh, nm, x0, x);
    d0_kernel<<<576, 256, 0, stream>>>(x0, d0b);

    for (int l = 0; l < NLAY; ++l) {
        radial_kernel<<<576, 256, 0, stream>>>(x, radb, cdiffb);
        for (int s = 0; s < NSUB; ++s) {
            const int g = l * NSUB + s;
            const float* We1 = gWe1 + (size_t)g * 258 * HDIM;
            // hi = h @ We1[0:128], hj = h @ We1[128:256]
            gemm_rows_kernel<<<96, 256, shGemm128, stream>>>(
                h, 128, 0, nullptr, 0, We1, nullptr,
                nullptr, nullptr, nullptr, 0, hi, nullptr);
            gemm_rows_kernel<<<96, 256, shGemm128, stream>>>(
                h, 128, 0, nullptr, 0, We1 + 128 * HDIM, nullptr,
                nullptr, nullptr, nullptr, 0, hj, nullptr);
            // fused pair MLP + masked j-reduction (j-split, accumulate)
            zero_kernel<<<768, 256, 0, stream>>>(agg, (int)NH);
            edge_fused_kernel<<<edgeGrid, 256, shEdge, stream>>>(
                hi, hj, radb, d0b, em, We1 + 256 * HDIM,
                gbe1 + (size_t)g * HDIM, gWe2 + (size_t)g * HDIM * HDIM,
                gbe2 + (size_t)g * HDIM, agg);
            // u = silu(h@Wn1a + agg@Wn1b + bn1)  -> bf16
            gemm_rows_kernel<<<96, 256, shGemm256, stream>>>(
                h, 128, 0, agg, 128,
                gWn1 + (size_t)g * 256 * HDIM, gWn1 + (size_t)g * 256 * HDIM + 128 * HDIM,
                gbn1 + (size_t)g * HDIM, nullptr, nullptr, 1, nullptr, ubf);
            // h = (h + u@Wn2 + bn2) * node_mask
            gemm_rows_kernel<<<96, 256, shGemm128, stream>>>(
                ubf, 128, 1, nullptr, 0, gWn2 + (size_t)g * HDIM * HDIM, nullptr,
                gbn2 + (size_t)g * HDIM, h, nm, 0, h, nullptr);
        }
        const float* Wc1 = eWc1 + (size_t)l * 258 * HDIM;
        gemm_rows_kernel<<<96, 256, shGemm128, stream>>>(
            h, 128, 0, nullptr, 0, Wc1, nullptr,
            nullptr, nullptr, nullptr, 0, hi, nullptr);
        gemm_rows_kernel<<<96, 256, shGemm128, stream>>>(
            h, 128, 0, nullptr, 0, Wc1 + 128 * HDIM, nullptr,
            nullptr, nullptr, nullptr, 0, hj, nullptr);
        zero_kernel<<<18, 256, 0, stream>>>(xdelta, BSZ * NPT * 3);
        coord_fused_kernel<<<edgeGrid, 256, shCoord, stream>>>(
            hi, hj, radb, d0b, em, cdiffb, Wc1 + 256 * HDIM,
            ebc1 + (size_t)l * HDIM, eWc2 + (size_t)l * HDIM * HDIM,
            ebc2 + (size_t)l * HDIM, eWc3 + (size_t)l * HDIM, xdelta);
        xupdate_kernel<<<18, 256, 0, stream>>>(xdelta, nm, x);
    }

    zero_kernel<<<1, 64, 0, stream>>>(velsum, 64);
    vel_kernel<<<6, 256, 0, stream>>>(x, x0, nm, vel, velsum, nsum);
    final_kernel<<<6, 256, 0, stream>>>(vel, velsum, nsum, h, W_out, b_out, nm,
                                        (float*)d_out);
    (void)in_sizes; (void)n_in; (void)out_size; (void)ws_size;
}
